// TransformerBlock_89086211653668
// MI455X (gfx1250) — matrix operand order repaired and checked by reading, not compiled, not run on hardware
//
#include <hip/hip_runtime.h>
#include <hip/hip_bf16.h>

// ---------------------------------------------------------------------------
// Transformer block (sliding-window attn + top-2 MoE SwiGLU) for gfx1250.
// All heavy math via v_wmma_f32_16x16x32_bf16 (wave32), fp32 accumulate.
// Weights: converted+transposed to bf16 ONCE per launch (cvtT_k).
// Activations: produced in bf16 (rmsnorm/attn/swiglu epilogues), so BOTH
// sides of every GEMM stage as pure dword copies -- zero conversion VALU in
// the k-loops. GEMMs are double-buffered LDS software pipelines.
// MoE is *routed* (top-2 only): ~103 GFLOP instead of the reference's 412.
// ---------------------------------------------------------------------------

#define HID 1024
#define NH  16
#define HD  64
#define WIN 512
#define DFF 4096
#define NE  8
#define TOPK 2
#define SEQ 2048
#define EPS 1e-6f

typedef __attribute__((ext_vector_type(16))) __bf16        v16bf;
typedef __attribute__((ext_vector_type(8)))  float         v8f;
typedef __attribute__((ext_vector_type(8)))  unsigned int  v8u;

// ---- bf16 helpers ---------------------------------------------------------
static __device__ __forceinline__ unsigned short f2bf(float f) {
  unsigned int u = __builtin_bit_cast(unsigned int, f);
  u += 0x7fffu + ((u >> 16) & 1u);   // round-to-nearest-even
  return (unsigned short)(u >> 16);
}
static __device__ __forceinline__ unsigned int pack2(float a, float b) {
#if __has_builtin(__builtin_amdgcn_cvt_pk_bf16_f32)
  auto r = __builtin_amdgcn_cvt_pk_bf16_f32(a, b);
  return __builtin_bit_cast(unsigned int, r);
#else
  return (unsigned int)f2bf(a) | ((unsigned int)f2bf(b) << 16);
#endif
}
static __device__ __forceinline__ float bf2f(unsigned short h) {
  unsigned int u = ((unsigned int)h) << 16;
  return __builtin_bit_cast(float, u);
}

// ---- WMMA wrapper ---------------------------------------------------------
static __device__ __forceinline__ v8f wmma_bf16(v8u a, v8u b, v8f c) {
  return __builtin_amdgcn_wmma_f32_16x16x32_bf16(
      false, __builtin_bit_cast(v16bf, a),
      false, __builtin_bit_cast(v16bf, b),
      (short)0, c, false, false);
}

// A-fragment K mapping (16-bit A 16x32, wave32):
// lanes 0-15: K in {0..7} u {16..23}; lanes 16-31: K in {8..15} u {24..31}
static __device__ __forceinline__ int kmapA(int v, int g) {
  return (v < 4) ? (8 * g + 2 * v) : (16 + 8 * g + 2 * (v - 4));
}

// Load A fragment from LDS (row-major [m][k], stride ld ushorts, 32-k tile)
static __device__ __forceinline__ v8u fragA(const unsigned short* S, int ld, int row0) {
  int lane = threadIdx.x & 31, g = lane >> 4, m = lane & 15;
  v8u a;
#pragma unroll
  for (int v = 0; v < 8; ++v)
    a[v] = *(const unsigned int*)&S[(row0 + m) * ld + kmapA(v, g)];
  return a;
}

// Load B fragment from transposed LDS (Bt[n][k], stride ld ushorts).
static __device__ __forceinline__ v8u fragB(const unsigned short* S, int ld, int n0, int k0) {
  int lane = threadIdx.x & 31, g = lane >> 4, n = lane & 15;
  v8u b;
#pragma unroll
  for (int v = 0; v < 8; ++v)
    b[v] = *(const unsigned int*)&S[(n0 + n) * ld + k0 + kmapA(v, g)];
  return b;
}

static __device__ __forceinline__ float rmax16(float v) {
#pragma unroll
  for (int m = 1; m < 16; m <<= 1) v = fmaxf(v, __shfl_xor(v, m, 32));
  return v;
}
static __device__ __forceinline__ float rsum16(float v) {
#pragma unroll
  for (int m = 1; m < 16; m <<= 1) v += __shfl_xor(v, m, 32);
  return v;
}

// ---------------------------------------------------------------------------
// One-time weight convert + transpose: src f32 [K][N] -> dst bf16 [N][K].
// grid: (N/64, K/64, batch)
// ---------------------------------------------------------------------------
__global__ __launch_bounds__(256) void cvtT_k(const float* __restrict__ src,
                                              unsigned short* __restrict__ dst,
                                              int K, int N) {
  src += (size_t)blockIdx.z * K * N;
  dst += (size_t)blockIdx.z * K * N;
  __shared__ float T[64][65];
  const int kt = blockIdx.y * 64, nt = blockIdx.x * 64;
  const int tid = threadIdx.x;
#pragma unroll
  for (int i = 0; i < 4; ++i) {
    int c = tid + 256 * i;                 // 64 rows x 16 float4
    int r = c >> 4, c4 = (c & 15) * 4;
    float4 v = *(const float4*)(src + (size_t)(kt + r) * N + nt + c4);
    T[r][c4 + 0] = v.x; T[r][c4 + 1] = v.y;
    T[r][c4 + 2] = v.z; T[r][c4 + 3] = v.w;
  }
  __syncthreads();
#pragma unroll
  for (int i = 0; i < 8; ++i) {
    int c = tid + 256 * i;                 // 64 n-rows x 32 packed dwords
    int n = c >> 5, kd = (c & 31) * 2;
    unsigned int d = pack2(T[kd][n], T[kd + 1][n]);
    *(unsigned int*)&dst[(size_t)(nt + n) * K + kt + kd] = d;
  }
}

// ---------------------------------------------------------------------------
// Generic 128x128 bf16-WMMA GEMM: A bf16 [M][K], B bf16 TRANSPOSED [N][K],
// C f32 [M][N]; optional residual add, per-expert dynamic M, row gather.
// Both stagings are pure dword copies. Double-buffered pipeline.
// grid: x = N/128, y = ceil(Mmax/128), z = experts (or 1)
// ---------------------------------------------------------------------------
__global__ __launch_bounds__(256) void gemm128(
    const unsigned short* __restrict__ A, int lda,
    const unsigned short* __restrict__ Bt, int ldb /* = K */, size_t bstride_e,
    float* __restrict__ C, int ldc, int M, int K,
    const float* __restrict__ resid,
    const int* __restrict__ counts, const int* __restrict__ offsets,
    const int* __restrict__ gather) {
  __shared__ unsigned short As[2][128 * 34];
  __shared__ unsigned short Bs[2][128 * 34];

  const int e = blockIdx.z;
  int base = 0, mcnt = M;
  if (counts) { base = offsets[e]; mcnt = counts[e]; Bt += bstride_e * (size_t)e; }
  const int mb = blockIdx.y, nb = blockIdx.x;
  if (mb * 128 >= mcnt) return;

  const int tid = threadIdx.x;
  const int w = tid >> 5, lane = tid & 31, g = lane >> 4, ln = lane & 15;
  const int wr = w >> 1, wc = w & 1;   // 4x2 wave grid -> 32x64 per wave

  // ---- staging geometry (pure bf16 copies; 16 contiguous bf16 per thread)
  const int trow = tid >> 1, tseg = (tid & 1) * 16;
  int rl = mb * 128 + trow;
  int ar = (rl < mcnt) ? rl : 0;
  int grow = gather ? gather[base + ar] : (base + ar);
  const unsigned short* aptr = A + (size_t)grow * lda + tseg;
  const unsigned short* bptr = Bt + (size_t)(nb * 128 + trow) * ldb + tseg;

  uint4 pa0, pa1, pb0, pb1;
  auto loadTiles = [&](int kt) {
    pa0 = *(const uint4*)(aptr + kt);
    pa1 = *(const uint4*)(aptr + kt + 8);
    pb0 = *(const uint4*)(bptr + kt);
    pb1 = *(const uint4*)(bptr + kt + 8);
  };
  auto storeTiles = [&](unsigned short* Ad, unsigned short* Bd) {
    unsigned int* ad = (unsigned int*)&Ad[trow * 34 + tseg];
    ad[0] = pa0.x; ad[1] = pa0.y; ad[2] = pa0.z; ad[3] = pa0.w;
    unsigned int* ad2 = (unsigned int*)&Ad[trow * 34 + tseg + 8];
    ad2[0] = pa1.x; ad2[1] = pa1.y; ad2[2] = pa1.z; ad2[3] = pa1.w;
    unsigned int* bd = (unsigned int*)&Bd[trow * 34 + tseg];
    bd[0] = pb0.x; bd[1] = pb0.y; bd[2] = pb0.z; bd[3] = pb0.w;
    unsigned int* bd2 = (unsigned int*)&Bd[trow * 34 + tseg + 8];
    bd2[0] = pb1.x; bd2[1] = pb1.y; bd2[2] = pb1.z; bd2[3] = pb1.w;
  };

  v8f zero = {};
  v8f acc[2][4];
#pragma unroll
  for (int mf = 0; mf < 2; ++mf)
#pragma unroll
    for (int nf = 0; nf < 4; ++nf) acc[mf][nf] = zero;

  const int ntiles = K / 32;
  loadTiles(0);
  storeTiles(As[0], Bs[0]);
  __syncthreads();

  for (int it = 0; it < ntiles; ++it) {
    const int buf = it & 1;
    const bool more = (it + 1) < ntiles;
    if (more) loadTiles((it + 1) * 32);   // global loads in flight over WMMAs

    v8u af[2], bf[4];
    af[0] = fragA(As[buf], 34, wr * 32);
    af[1] = fragA(As[buf], 34, wr * 32 + 16);
#pragma unroll
    for (int nf = 0; nf < 4; ++nf) bf[nf] = fragB(Bs[buf], 34, wc * 64 + nf * 16, 0);
#pragma unroll
    for (int mf = 0; mf < 2; ++mf)
#pragma unroll
      for (int nf = 0; nf < 4; ++nf)
        acc[mf][nf] = wmma_bf16(af[mf], bf[nf], acc[mf][nf]);

    if (more) storeTiles(As[buf ^ 1], Bs[buf ^ 1]);   // waits loadcnt here
    __syncthreads();
  }

  // ---- epilogue (masked rows for dynamic-M experts) ----
#pragma unroll
  for (int mf = 0; mf < 2; ++mf)
#pragma unroll
    for (int nf = 0; nf < 4; ++nf)
#pragma unroll
      for (int j = 0; j < 8; ++j) {
        int r = mb * 128 + wr * 32 + mf * 16 + j + 8 * g;
        if (r < mcnt) {
          int col = nb * 128 + wc * 64 + nf * 16 + ln;
          size_t off = (size_t)(base + r) * ldc + col;
          float v = acc[mf][nf][j];
          if (resid) v += resid[off];
          C[off] = v;
        }
      }
}

// ---------------------------------------------------------------------------
// Fused SwiGLU expert GEMM: H = silu(Yg @ W1_e) * (Yg @ W3_e), bf16 out.
// A bf16 gathered rows, W1t/W3t bf16 transposed [DFF][HID] per expert.
// tile 128x64, masked stores, double-buffered. grid (64,16,8).
// ---------------------------------------------------------------------------
__global__ __launch_bounds__(256) void swiglu_gemm(
    const unsigned short* __restrict__ Y, const unsigned short* __restrict__ W1t,
    const unsigned short* __restrict__ W3t, unsigned short* __restrict__ H,
    const int* __restrict__ counts, const int* __restrict__ offsets,
    const int* __restrict__ token_of) {
  __shared__ unsigned short As[2][128 * 34];
  __shared__ unsigned short B1s[2][64 * 34];
  __shared__ unsigned short B3s[2][64 * 34];

  const int e = blockIdx.z;
  const int base = offsets[e], mcnt = counts[e];
  const int mb = blockIdx.y, nb = blockIdx.x;
  if (mb * 128 >= mcnt) return;
  const unsigned short* W1e = W1t + (size_t)e * HID * DFF;
  const unsigned short* W3e = W3t + (size_t)e * HID * DFF;

  const int tid = threadIdx.x;
  const int w = tid >> 5, lane = tid & 31, g = lane >> 4, ln = lane & 15;
  const int wr = w >> 1, wc = w & 1;   // 4x2 waves -> 32x32 per wave, 2 outputs

  // A: 128 rows x 32 k; 16 contiguous bf16 per thread (gathered rows)
  const int trow = tid >> 1, tseg = (tid & 1) * 16;
  int rl = mb * 128 + trow;
  int ar = (rl < mcnt) ? rl : 0;
  const unsigned short* aptr = Y + (size_t)token_of[base + ar] * HID + tseg;
  // B: 64 n-rows x 32 k; 8 contiguous bf16 (uint4) per thread
  const int brow = tid >> 2, bseg = (tid & 3) * 8;
  const unsigned short* b1p = W1e + (size_t)(nb * 64 + brow) * HID + bseg;
  const unsigned short* b3p = W3e + (size_t)(nb * 64 + brow) * HID + bseg;

  uint4 pa0, pa1, p1, p3;
  auto loadTiles = [&](int kt) {
    pa0 = *(const uint4*)(aptr + kt);
    pa1 = *(const uint4*)(aptr + kt + 8);
    p1 = *(const uint4*)(b1p + kt);
    p3 = *(const uint4*)(b3p + kt);
  };
  auto storeTiles = [&](unsigned short* Ad, unsigned short* B1d, unsigned short* B3d) {
    unsigned int* ad = (unsigned int*)&Ad[trow * 34 + tseg];
    ad[0] = pa0.x; ad[1] = pa0.y; ad[2] = pa0.z; ad[3] = pa0.w;
    unsigned int* ad2 = (unsigned int*)&Ad[trow * 34 + tseg + 8];
    ad2[0] = pa1.x; ad2[1] = pa1.y; ad2[2] = pa1.z; ad2[3] = pa1.w;
    unsigned int* d1 = (unsigned int*)&B1d[brow * 34 + bseg];
    d1[0] = p1.x; d1[1] = p1.y; d1[2] = p1.z; d1[3] = p1.w;
    unsigned int* d3 = (unsigned int*)&B3d[brow * 34 + bseg];
    d3[0] = p3.x; d3[1] = p3.y; d3[2] = p3.z; d3[3] = p3.w;
  };

  v8f zero = {};
  v8f a1[2][2], a3[2][2];
#pragma unroll
  for (int mf = 0; mf < 2; ++mf)
#pragma unroll
    for (int nf = 0; nf < 2; ++nf) { a1[mf][nf] = zero; a3[mf][nf] = zero; }

  const int ntiles = HID / 32;
  loadTiles(0);
  storeTiles(As[0], B1s[0], B3s[0]);
  __syncthreads();

  for (int it = 0; it < ntiles; ++it) {
    const int buf = it & 1;
    const bool more = (it + 1) < ntiles;
    if (more) loadTiles((it + 1) * 32);

    v8u af[2], b1f[2], b3f[2];
    af[0] = fragA(As[buf], 34, wr * 32);
    af[1] = fragA(As[buf], 34, wr * 32 + 16);
#pragma unroll
    for (int nf = 0; nf < 2; ++nf) {
      b1f[nf] = fragB(B1s[buf], 34, wc * 32 + nf * 16, 0);
      b3f[nf] = fragB(B3s[buf], 34, wc * 32 + nf * 16, 0);
    }
#pragma unroll
    for (int mf = 0; mf < 2; ++mf)
#pragma unroll
      for (int nf = 0; nf < 2; ++nf) {
        a1[mf][nf] = wmma_bf16(af[mf], b1f[nf], a1[mf][nf]);
        a3[mf][nf] = wmma_bf16(af[mf], b3f[nf], a3[mf][nf]);
      }

    if (more) storeTiles(As[buf ^ 1], B1s[buf ^ 1], B3s[buf ^ 1]);
    __syncthreads();
  }

#pragma unroll
  for (int mf = 0; mf < 2; ++mf)
#pragma unroll
    for (int nf = 0; nf < 2; ++nf)
#pragma unroll
      for (int j = 0; j < 8; ++j) {
        int r = mb * 128 + wr * 32 + mf * 16 + j + 8 * g;
        if (r < mcnt) {
          int col = nb * 64 + wc * 32 + nf * 16 + ln;
          float x1v = a1[mf][nf][j];
          float sw = x1v / (1.f + __expf(-x1v));
          H[(size_t)(base + r) * DFF + col] = f2bf(sw * a3[mf][nf][j]);
        }
      }
}

// ---------------------------------------------------------------------------
// Sliding-window flash attention. grid (qblocks=16, heads=16), 256 threads.
// Consumes f32 q/k/v (post-RoPE); emits bf16 attn for the out-proj GEMM.
// ---------------------------------------------------------------------------
__global__ __launch_bounds__(256) void attn_kernel(
    const float* __restrict__ qf, const float* __restrict__ kf,
    const float* __restrict__ vf, unsigned short* __restrict__ attn) {
  const int qb = blockIdx.x, h = blockIdx.y;
  const int tid = threadIdx.x;
  const int w = tid >> 5, lane = tid & 31, g = lane >> 4, ln = lane & 15;

  __shared__ unsigned short Ks[32 * 66];    // k tile [key][d], padded
  __shared__ unsigned short Vt[64 * 34];    // v tile transposed [d][key]
  __shared__ unsigned short Ps[8 * 16 * 34];// per-wave P buffers
  unsigned short* Pw = &Ps[w * 16 * 34];

  const int q0 = qb * 128 + w * 16;

  v8u aq[2];
  {
    const float* qrow = qf + (size_t)(q0 + ln) * HID + h * 64;
#pragma unroll
    for (int c = 0; c < 2; ++c)
#pragma unroll
      for (int v = 0; v < 8; ++v) {
        int k = c * 32 + kmapA(v, g);
        aq[c][v] = pack2(qrow[k], qrow[k + 1]);
      }
  }

  v8f zero = {};
  v8f o[4];
#pragma unroll
  for (int f = 0; f < 4; ++f) o[f] = zero;
  float mst[8], lst[8];
#pragma unroll
  for (int j = 0; j < 8; ++j) { mst[j] = -3.0e38f; lst[j] = 0.f; }

  int ks0 = qb * 128 - (WIN - 1);
  ks0 = (ks0 < 0) ? 0 : (ks0 & ~31);
  const int ksend = qb * 128 + 127;

  for (int ks = ks0; ks <= ksend; ks += 32) {
    {
      int r = tid >> 3, c0 = (tid & 7) * 8;
      const float* kp = kf + (size_t)(ks + r) * HID + h * 64 + c0;
      const float* vp = vf + (size_t)(ks + r) * HID + h * 64 + c0;
      float4 ka = *(const float4*)kp, kb2 = *(const float4*)(kp + 4);
      float4 va = *(const float4*)vp, vb2 = *(const float4*)(vp + 4);
      unsigned int* kd = (unsigned int*)&Ks[r * 66 + c0];
      kd[0] = pack2(ka.x, ka.y); kd[1] = pack2(ka.z, ka.w);
      kd[2] = pack2(kb2.x, kb2.y); kd[3] = pack2(kb2.z, kb2.w);
      Vt[(c0 + 0) * 34 + r] = f2bf(va.x); Vt[(c0 + 1) * 34 + r] = f2bf(va.y);
      Vt[(c0 + 2) * 34 + r] = f2bf(va.z); Vt[(c0 + 3) * 34 + r] = f2bf(va.w);
      Vt[(c0 + 4) * 34 + r] = f2bf(vb2.x); Vt[(c0 + 5) * 34 + r] = f2bf(vb2.y);
      Vt[(c0 + 6) * 34 + r] = f2bf(vb2.z); Vt[(c0 + 7) * 34 + r] = f2bf(vb2.w);
    }
    __syncthreads();

    v8f sc[2];
    sc[0] = zero; sc[1] = zero;
#pragma unroll
    for (int nf = 0; nf < 2; ++nf)
#pragma unroll
      for (int ch = 0; ch < 2; ++ch)
        sc[nf] = wmma_bf16(aq[ch], fragB(Ks, 66, nf * 16, ch * 32), sc[nf]);

#pragma unroll
    for (int nf = 0; nf < 2; ++nf)
#pragma unroll
      for (int j = 0; j < 8; ++j) {
        int key = ks + nf * 16 + ln;
        int qr = q0 + j + 8 * g;
        float sv = sc[nf][j] * 0.125f;
        bool ok = (key <= qr) && (qr - key < WIN);
        sc[nf][j] = ok ? sv : -3.0e38f;
      }
    float alpha[8];
#pragma unroll
    for (int j = 0; j < 8; ++j) {
      float mx = rmax16(fmaxf(sc[0][j], sc[1][j]));
      float mnew = fmaxf(mst[j], mx);
      float p0 = (sc[0][j] <= -1e30f) ? 0.f : __expf(sc[0][j] - mnew);
      float p1 = (sc[1][j] <= -1e30f) ? 0.f : __expf(sc[1][j] - mnew);
      float a = __expf(mst[j] - mnew);
      lst[j] = lst[j] * a + rsum16(p0 + p1);
      mst[j] = mnew;
      alpha[j] = a;
      sc[0][j] = p0; sc[1][j] = p1;
    }
#pragma unroll
    for (int f = 0; f < 4; ++f)
#pragma unroll
      for (int j = 0; j < 8; ++j) o[f][j] *= alpha[j];

#pragma unroll
    for (int nf = 0; nf < 2; ++nf)
#pragma unroll
      for (int j = 0; j < 8; ++j)
        Pw[(j + 8 * g) * 34 + nf * 16 + ln] = f2bf(sc[nf][j]);

    v8u ap = fragA(Pw, 34, 0);
#pragma unroll
    for (int f = 0; f < 4; ++f)
      o[f] = wmma_bf16(ap, fragB(Vt, 34, f * 16, 0), o[f]);
    __syncthreads();
  }

#pragma unroll
  for (int f = 0; f < 4; ++f)
#pragma unroll
    for (int j = 0; j < 8; ++j) {
      int row = q0 + j + 8 * g;
      attn[(size_t)row * HID + h * 64 + f * 16 + ln] = f2bf(o[f][j] / lst[j]);
    }
}

// ---------------------------------------------------------------------------
// Small elementwise / routing kernels
// ---------------------------------------------------------------------------
__global__ __launch_bounds__(256) void rmsnorm_k(const float* __restrict__ x,
                                                 const float* __restrict__ wgt,
                                                 unsigned short* __restrict__ yb) {
  const int t = blockIdx.x, tid = threadIdx.x;
  const float* xp = x + (size_t)t * HID;
  float4 xv = *(const float4*)(xp + tid * 4);
  float s = xv.x * xv.x + xv.y * xv.y + xv.z * xv.z + xv.w * xv.w;
#pragma unroll
  for (int m = 16; m >= 1; m >>= 1) s += __shfl_xor(s, m, 32);
  __shared__ float red[8];
  __shared__ float rs_sh;
  if ((tid & 31) == 0) red[tid >> 5] = s;
  __syncthreads();
  if (tid == 0) {
    float tot = 0.f;
#pragma unroll
    for (int i = 0; i < 8; ++i) tot += red[i];
    rs_sh = rsqrtf(tot / (float)HID + EPS);
  }
  __syncthreads();
  float rs = rs_sh;
  float4 wv = *(const float4*)(wgt + tid * 4);
  unsigned int d0 = pack2(xv.x * rs * wv.x, xv.y * rs * wv.y);
  unsigned int d1 = pack2(xv.z * rs * wv.z, xv.w * rs * wv.w);
  *(uint2*)&yb[(size_t)t * HID + tid * 4] = make_uint2(d0, d1);
}

__global__ void rope_k(float* q, float* k, const float* __restrict__ R) {
  int idx = blockIdx.x * blockDim.x + threadIdx.x;
  if (idx >= SEQ * NH * 32) return;
  int i = idx & 31;
  int h = (idx >> 5) & 15;
  int s = idx >> 9;
  float c  = R[((size_t)s * 64 + 2 * i) * 64 + 2 * i];
  float sn = R[((size_t)s * 64 + 2 * i + 1) * 64 + 2 * i];
  size_t o = (size_t)s * HID + h * 64 + 2 * i;
  float q0 = q[o], q1 = q[o + 1];
  q[o] = c * q0 + sn * q1;  q[o + 1] = -sn * q0 + c * q1;
  float k0 = k[o], k1 = k[o + 1];
  k[o] = c * k0 + sn * k1;  k[o + 1] = -sn * k0 + c * k1;
}

__global__ void init_k(int* counts, int* cursor) {
  if (threadIdx.x < NE) { counts[threadIdx.x] = 0; cursor[threadIdx.x] = 0; }
}

__global__ void route_k(const unsigned short* __restrict__ y2b,
                        const float* __restrict__ gw,
                        int* __restrict__ ti, float* __restrict__ gv,
                        int* __restrict__ counts) {
  int t = blockIdx.x * 256 + threadIdx.x;
  if (t >= SEQ) return;
  float acc[NE];
#pragma unroll
  for (int e = 0; e < NE; ++e) acc[e] = 0.f;
  const unsigned short* yp = y2b + (size_t)t * HID;
  for (int i = 0; i < HID; ++i) {
    float xv = bf2f(yp[i]);
#pragma unroll
    for (int e = 0; e < NE; ++e) acc[e] += xv * gw[i * NE + e];
  }
  int i0 = 0; float v0 = acc[0];
#pragma unroll
  for (int e = 1; e < NE; ++e) if (acc[e] > v0) { v0 = acc[e]; i0 = e; }
  int i1 = (i0 == 0) ? 1 : 0; float v1 = acc[i1];
#pragma unroll
  for (int e = 0; e < NE; ++e)
    if (e != i0 && acc[e] > v1) { v1 = acc[e]; i1 = e; }
  float e1 = __expf(v1 - v0);
  float inv = 1.f / (1.f + e1);
  ti[t * 2] = i0; ti[t * 2 + 1] = i1;
  gv[t * 2] = inv; gv[t * 2 + 1] = e1 * inv;
  atomicAdd(&counts[i0], 1);
  atomicAdd(&counts[i1], 1);
}

__global__ void scan_k(const int* counts, int* offsets, int* cursor) {
  if (threadIdx.x == 0) {
    int o = 0;
    for (int e = 0; e < NE; ++e) { offsets[e] = o; cursor[e] = o; o += counts[e]; }
  }
}

__global__ void scatter_k(const int* __restrict__ ti, int* cursor,
                          int* __restrict__ token_of, int* __restrict__ slot_of) {
  int t = blockIdx.x * 256 + threadIdx.x;
  if (t >= SEQ) return;
#pragma unroll
  for (int k = 0; k < TOPK; ++k) {
    int e = ti[t * 2 + k];
    int pos = atomicAdd(&cursor[e], 1);
    token_of[pos] = t;
    slot_of[t * 2 + k] = pos;
  }
}

__global__ void combine_k(const float* __restrict__ x1, const float* __restrict__ out2,
                          const float* __restrict__ gv, const int* __restrict__ slot_of,
                          float* __restrict__ out) {
  int idx = blockIdx.x * 256 + threadIdx.x;
  if (idx >= SEQ * HID) return;
  int t = idx >> 10, d = idx & (HID - 1);
  float v = x1[idx];
  v += gv[t * 2]     * out2[(size_t)slot_of[t * 2]     * HID + d];
  v += gv[t * 2 + 1] * out2[(size_t)slot_of[t * 2 + 1] * HID + d];
  out[idx] = v;
}

// ---------------------------------------------------------------------------
extern "C" void kernel_launch(void* const* d_in, const int* in_sizes, int n_in,
                              void* d_out, int out_size, void* d_ws, size_t ws_size,
                              hipStream_t stream) {
  const float* x    = (const float*)d_in[0];
  const float* R    = (const float*)d_in[1];
  const float* r1w  = (const float*)d_in[2];
  const float* wq   = (const float*)d_in[3];
  const float* wk   = (const float*)d_in[4];
  const float* wv   = (const float*)d_in[5];
  const float* wo   = (const float*)d_in[6];
  const float* r2w  = (const float*)d_in[7];
  const float* gw   = (const float*)d_in[8];
  const float* w1   = (const float*)d_in[9];
  const float* w2   = (const float*)d_in[10];
  const float* w3   = (const float*)d_in[11];
  float* out = (float*)d_out;

  // ---- carve workspace (~300 MB) ----
  char* p = (char*)d_ws;
  auto carve = [&](size_t bytes) { void* r = (void*)p; p += (bytes + 255) & ~(size_t)255; return r; };
  unsigned short* y    = (unsigned short*)carve((size_t)SEQ * HID * 2);
  float* qf   = (float*)carve((size_t)SEQ * HID * 4);
  float* kf   = (float*)carve((size_t)SEQ * HID * 4);
  float* vf   = (float*)carve((size_t)SEQ * HID * 4);
  unsigned short* attn = (unsigned short*)carve((size_t)SEQ * HID * 2);
  float* x1   = (float*)carve((size_t)SEQ * HID * 4);
  unsigned short* y2 = (unsigned short*)carve((size_t)SEQ * HID * 2);
  unsigned short* H  = (unsigned short*)carve((size_t)SEQ * TOPK * DFF * 2);
  float* out2 = (float*)carve((size_t)SEQ * TOPK * HID * 4);
  unsigned short* wqt = (unsigned short*)carve((size_t)HID * HID * 2);
  unsigned short* wkt = (unsigned short*)carve((size_t)HID * HID * 2);
  unsigned short* wvt = (unsigned short*)carve((size_t)HID * HID * 2);
  unsigned short* wot = (unsigned short*)carve((size_t)HID * HID * 2);
  unsigned short* w1t = (unsigned short*)carve((size_t)NE * HID * DFF * 2);
  unsigned short* w3t = (unsigned short*)carve((size_t)NE * HID * DFF * 2);
  unsigned short* w2t = (unsigned short*)carve((size_t)NE * DFF * HID * 2);
  int* counts   = (int*)carve(NE * 4);
  int* offsets  = (int*)carve(NE * 4);
  int* cursor   = (int*)carve(NE * 4);
  int* token_of = (int*)carve((size_t)SEQ * TOPK * 4);
  int* slot_of  = (int*)carve((size_t)SEQ * TOPK * 4);
  int* ti       = (int*)carve((size_t)SEQ * TOPK * 4);
  float* gv     = (float*)carve((size_t)SEQ * TOPK * 4);
  (void)in_sizes; (void)n_in; (void)out_size; (void)ws_size;

  init_k<<<1, 32, 0, stream>>>(counts, cursor);

  // ---- one-time weight convert+transpose to bf16 [N][K] ----
  cvtT_k<<<dim3(HID / 64, HID / 64, 1), 256, 0, stream>>>(wq, wqt, HID, HID);
  cvtT_k<<<dim3(HID / 64, HID / 64, 1), 256, 0, stream>>>(wk, wkt, HID, HID);
  cvtT_k<<<dim3(HID / 64, HID / 64, 1), 256, 0, stream>>>(wv, wvt, HID, HID);
  cvtT_k<<<dim3(HID / 64, HID / 64, 1), 256, 0, stream>>>(wo, wot, HID, HID);
  cvtT_k<<<dim3(DFF / 64, HID / 64, NE), 256, 0, stream>>>(w1, w1t, HID, DFF);
  cvtT_k<<<dim3(DFF / 64, HID / 64, NE), 256, 0, stream>>>(w3, w3t, HID, DFF);
  cvtT_k<<<dim3(HID / 64, DFF / 64, NE), 256, 0, stream>>>(w2, w2t, DFF, HID);

  // attention branch
  rmsnorm_k<<<SEQ, 256, 0, stream>>>(x, r1w, y);
  gemm128<<<dim3(HID / 128, SEQ / 128, 1), 256, 0, stream>>>(
      y, HID, wqt, HID, 0, qf, HID, SEQ, HID, nullptr, nullptr, nullptr, nullptr);
  gemm128<<<dim3(HID / 128, SEQ / 128, 1), 256, 0, stream>>>(
      y, HID, wkt, HID, 0, kf, HID, SEQ, HID, nullptr, nullptr, nullptr, nullptr);
  gemm128<<<dim3(HID / 128, SEQ / 128, 1), 256, 0, stream>>>(
      y, HID, wvt, HID, 0, vf, HID, SEQ, HID, nullptr, nullptr, nullptr, nullptr);
  rope_k<<<(SEQ * NH * 32 + 255) / 256, 256, 0, stream>>>(qf, kf, R);
  attn_kernel<<<dim3(SEQ / 128, NH), 256, 0, stream>>>(qf, kf, vf, attn);
  gemm128<<<dim3(HID / 128, SEQ / 128, 1), 256, 0, stream>>>(
      attn, HID, wot, HID, 0, x1, HID, SEQ, HID, x, nullptr, nullptr, nullptr);

  // MoE branch (routed top-2)
  rmsnorm_k<<<SEQ, 256, 0, stream>>>(x1, r2w, y2);
  route_k<<<SEQ / 256, 256, 0, stream>>>(y2, gw, ti, gv, counts);
  scan_k<<<1, 32, 0, stream>>>(counts, offsets, cursor);
  scatter_k<<<SEQ / 256, 256, 0, stream>>>(ti, cursor, token_of, slot_of);
  swiglu_gemm<<<dim3(DFF / 64, SEQ / 128, NE), 256, 0, stream>>>(
      y2, w1t, w3t, H, counts, offsets, token_of);
  gemm128<<<dim3(HID / 128, SEQ / 128, NE), 256, 0, stream>>>(
      H, DFF, w2t, DFF, (size_t)DFF * HID, out2, HID, 0, DFF,
      nullptr, counts, offsets, nullptr);
  combine_k<<<(SEQ * HID + 255) / 256, 256, 0, stream>>>(x1, out2, gv, slot_of, out);
}
